// GraphSAGE_11235634446655
// MI455X (gfx1250) — matrix operand order repaired and checked by reading, not compiled, not run on hardware
//
#include <hip/hip_runtime.h>

typedef __attribute__((ext_vector_type(16))) __bf16 v16bf;
typedef __attribute__((ext_vector_type(4)))  __bf16 v4bf;
typedef __attribute__((ext_vector_type(8)))  float  v8f;
typedef __attribute__((ext_vector_type(4)))  float  f4;
typedef __attribute__((ext_vector_type(4)))  unsigned int u32x4;
typedef __attribute__((ext_vector_type(8)))  unsigned int u32x8;

// Hardware fp32 atomic add at device scope (RMW executes in L2; avoids any
// CAS-loop lowering of the HIP atomic builtins).
__device__ __forceinline__ void atom_add_f32(float* p, float v) {
    unsigned long long a = (unsigned long long)p;
    asm volatile("global_atomic_add_f32 %0, %1, off scope:SCOPE_DEV"
                 :: "v"(a), "v"(v) : "memory");
}

// ---------------------------------------------------------------------------
// fp32 -> bf16 conversion (4 elements / thread)
// ---------------------------------------------------------------------------
__global__ __launch_bounds__(256)
void k_cvt_bf16(const float* __restrict__ in, __bf16* __restrict__ out, int n4) {
    int t = blockIdx.x * 256 + threadIdx.x;
    if (t >= n4) return;
    f4 v = ((const f4*)in)[t];
    v4bf o;
    o.x = (__bf16)v.x; o.y = (__bf16)v.y; o.z = (__bf16)v.z; o.w = (__bf16)v.w;
    ((v4bf*)out)[t] = o;
}

// Pack [W_top ; W_bot] (each rows x 256 fp32, row-major) into one bf16 matrix.
__global__ __launch_bounds__(256)
void k_pack_w(const float* __restrict__ Wtop, const float* __restrict__ Wbot,
              int rows_top, int rows_total, __bf16* __restrict__ wc) {
    int t = blockIdx.x * 256 + threadIdx.x;          // one thread per 4 elems
    int total4 = rows_total * 64;
    if (t >= total4) return;
    int row = t >> 6, c = (t & 63) << 2;
    const float* s = (row < rows_top) ? (Wtop + (size_t)row * 256)
                                      : (Wbot + (size_t)(row - rows_top) * 256);
    f4 v = *(const f4*)(s + c);
    v4bf o;
    o.x = (__bf16)v.x; o.y = (__bf16)v.y; o.z = (__bf16)v.z; o.w = (__bf16)v.w;
    *(v4bf*)(wc + (size_t)row * 256 + c) = o;
}

// ---------------------------------------------------------------------------
// degree[dst] += 1
// ---------------------------------------------------------------------------
__global__ __launch_bounds__(256)
void k_degree(const long long* __restrict__ dst, float* __restrict__ deg, int E) {
    int t = blockIdx.x * 256 + threadIdx.x;
    if (t >= E) return;
    atom_add_f32(&deg[(int)dst[t]], 1.0f);
}

// ---------------------------------------------------------------------------
// C[M x ldc] = A[M x 256](bf16) * B[ldc x 256](bf16)^T   via v_wmma bf16
// Block: 128 threads = 4 waves; block tile 64(M) x 64(N); K = 256.
// The 64x256 bf16 weight panel is DMA'd into LDS by the Tensor Data Mover,
// with TDM row padding (16B per 512B row) for ds_load bank-conflict avoidance.
// ---------------------------------------------------------------------------
__global__ __launch_bounds__(128)
void k_gemm_bf16(const __bf16* __restrict__ A, const __bf16* __restrict__ B,
                 float* __restrict__ C, int M, int ldc) {
    constexpr int BPAD = 264;                        // 256 + 8 bf16 row pad
    __shared__ __align__(16) __bf16 Bs[64 * BPAD];

    const int tid  = threadIdx.x;
    const int wave = tid >> 5;
    const int lane = tid & 31;
    const int m0 = blockIdx.x * 64 + wave * 16;
    const int n0 = blockIdx.y * 64;

    // --- TDM: tensor_load_to_lds of the 64x256 bf16 panel (one wave issues) ---
    if (wave == 0) {
        unsigned long long ga =
            (unsigned long long)(const void*)(B + (size_t)n0 * 256);
        u32x4 g0 = {};
        g0[0] = 1u;                                   // count=1 (valid D#)
        g0[1] = (unsigned)(size_t)(void*)Bs;          // lds_addr (bytes)
        g0[2] = (unsigned)(ga & 0xFFFFFFFFu);         // global_addr[31:0]
        g0[3] = (unsigned)((ga >> 32) & 0x01FFFFFFu)  // global_addr[56:32]
                | 0x80000000u;                        // type=2 ("image")
        u32x8 g1 = {};
        g1[0] = (1u << 16)        // data_size = 2 bytes
              | (1u << 20)        // pad_enable
              | (6u << 22)        // pad_interval: every 128 DWORDs (512B row)
              | (3u << 25);       // pad_amount: 4 DWORDs (16B)
        g1[1] = (256u & 0xFFFFu) << 16;  // tensor_dim0[15:0]=256 (elements)
        g1[2] = (64u << 16);             // tensor_dim1[15:0]=64
        g1[3] = (256u << 16);            // tile_dim0=256
        g1[4] = 64u;                     // tile_dim1=64, tile_dim2=0
        g1[5] = 256u;                    // tensor_dim0_stride=256 (elements)
        g1[6] = 0u;
        g1[7] = 0u;
        asm volatile("tensor_load_to_lds %0, %1" :: "s"(g0), "s"(g1) : "memory");
        __builtin_amdgcn_s_wait_tensorcnt(0);
    }
    __syncthreads();

    v8f acc[4] = {};

    // A fragment addressing per 16-bit A layout:
    //   lanes 0-15:  row m, K = {k..k+7, k+16..k+23}
    //   lanes 16-31: row m, K = {k+8..k+15, k+24..k+31}
    const int arow  = m0 + (lane & 15);
    const int rclmp = arow < M ? arow : (M - 1);
    const __bf16* ap = A + (size_t)rclmp * 256 + ((lane >> 4) << 3);

    // B fragment: lane n -> column n of tile; K = {k..k+7, k+16..k+23} / {k+8..k+15, k+24..k+31}
    const int bcol = lane & 15;
    const int bko  = (lane >> 4) << 3;

    union AB { v16bf v; f4 f[2]; };

    #pragma unroll
    for (int k = 0; k < 256; k += 32) {
        AB a;
        a.f[0] = *(const f4*)(ap + k);
        a.f[1] = *(const f4*)(ap + k + 16);
        #pragma unroll
        for (int j = 0; j < 4; ++j) {
            const __bf16* bp = Bs + (size_t)(j * 16 + bcol) * BPAD + k + bko;
            AB b;
            b.f[0] = *(const f4*)bp;
            b.f[1] = *(const f4*)(bp + 16);
            acc[j] = __builtin_amdgcn_wmma_f32_16x16x32_bf16(
                false, a.v, false, b.v, (short)0, acc[j], false, false);
        }
    }

    // D layout: lane n holds column n; VGPR r = row r (lanes 0-15) / r+8 (16-31)
    const int ccol  = n0 + (lane & 15);
    const int rbase = m0 + ((lane >> 4) << 3);
    #pragma unroll
    for (int j = 0; j < 4; ++j) {
        #pragma unroll
        for (int r = 0; r < 8; ++r) {
            int rr = rbase + r;
            if (rr < M) C[(size_t)rr * ldc + ccol + j * 16] = acc[j][r];
        }
    }
}

// ---------------------------------------------------------------------------
// One wave per edge: agg[dst][0:CH] += Y[src][0:CH]  (coalesced fp32 atomics)
// ---------------------------------------------------------------------------
template <int CH>
__global__ __launch_bounds__(256)
void k_scatter(const float* __restrict__ Y, int ldy,
               const long long* __restrict__ src, const long long* __restrict__ dst,
               float* __restrict__ agg, int E) {
    int w    = (blockIdx.x * 256 + threadIdx.x) >> 5;
    int lane = threadIdx.x & 31;
    if (w >= E) return;
    int s = (int)src[w], d = (int)dst[w];
    const float* y = Y + (size_t)s * ldy;
    float* a = agg + (size_t)d * CH;
    __builtin_prefetch(y, 0, 0);                     // global_prefetch_b8
    #pragma unroll
    for (int i = 0; i < CH / 32; ++i) {
        int c = lane + 32 * i;
        atom_add_f32(a + c, y[c]);
    }
}

// ---------------------------------------------------------------------------
// h = relu(agg/deg + b + Z), emitted directly as bf16 for the layer-2 GEMM
// ---------------------------------------------------------------------------
__global__ __launch_bounds__(256)
void k_finalize1(const float* __restrict__ agg, const float* __restrict__ deg,
                 const float* __restrict__ b, const float* __restrict__ Cz,
                 __bf16* __restrict__ h, int Nn) {
    int t = blockIdx.x * 256 + threadIdx.x;          // over Nn*64 (4 ch each)
    if (t >= Nn * 64) return;
    int n = t >> 6, c = (t & 63) << 2;
    float rd = 1.0f / fmaxf(deg[n], 1.0f);
    f4 a = *(const f4*)(agg + (size_t)n * 256 + c);
    f4 bb = *(const f4*)(b + c);
    f4 z = *(const f4*)(Cz + (size_t)n * 512 + c);   // Z part of [Y|Z], ld=512
    v4bf o;
    #pragma unroll
    for (int i = 0; i < 4; ++i) {
        float v = fmaf(a[i], rd, bb[i]) + z[i];
        v = fmaxf(v, 0.0f);
        o[i] = (__bf16)v;
    }
    *(v4bf*)(h + (size_t)n * 256 + c) = o;
}

// out = agg/deg + b + Z  (fp32, 64 channels)
__global__ __launch_bounds__(256)
void k_finalize2(const float* __restrict__ agg, const float* __restrict__ deg,
                 const float* __restrict__ b, const float* __restrict__ Cz,
                 float* __restrict__ out, int Nn) {
    int t = blockIdx.x * 256 + threadIdx.x;          // over Nn*16 (4 ch each)
    if (t >= Nn * 16) return;
    int n = t >> 4, c = (t & 15) << 2;
    float rd = 1.0f / fmaxf(deg[n], 1.0f);
    f4 a = *(const f4*)(agg + (size_t)n * 64 + c);
    f4 bb = *(const f4*)(b + c);
    f4 z = *(const f4*)(Cz + (size_t)n * 128 + c);   // Z2 part of [Y2|Z2], ld=128
    f4 o;
    #pragma unroll
    for (int i = 0; i < 4; ++i) o[i] = fmaf(a[i], rd, bb[i]) + z[i];
    *(f4*)(out + (size_t)n * 64 + c) = o;
}

// ---------------------------------------------------------------------------
extern "C" void kernel_launch(void* const* d_in, const int* in_sizes, int n_in,
                              void* d_out, int out_size, void* d_ws, size_t ws_size,
                              hipStream_t stream) {
    constexpr int NN = 50000, E = 800000;
    const float*     x    = (const float*)d_in[0];
    const float*     Wl1  = (const float*)d_in[1];
    const float*     b1   = (const float*)d_in[2];
    const float*     Wr1  = (const float*)d_in[3];
    const float*     Wl2  = (const float*)d_in[4];
    const float*     b2   = (const float*)d_in[5];
    const float*     Wr2  = (const float*)d_in[6];
    const long long* ei   = (const long long*)d_in[7];   // int64 edge_index [2,E]
    const long long* esrc = ei;
    const long long* edst = ei + E;
    float* out = (float*)d_out;
    (void)in_sizes; (void)n_in; (void)out_size; (void)ws_size;

    // workspace carve (~180 MB, with reuse)
    auto al = [](size_t v) { return (v + 255) & ~(size_t)255; };
    char* ws = (char*)d_ws;
    size_t off = 0;
    __bf16* xbf = (__bf16*)(ws + off); off += al((size_t)NN * 256 * 2); // X bf16, reused as H bf16
    __bf16* wcb = (__bf16*)(ws + off); off += al((size_t)512 * 256 * 2); // packed weights bf16
    float*  cbuf = (float*)(ws + off); off += al((size_t)NN * 512 * 4);  // [Y|Z] layer1, reused layer2
    float*  agg  = (float*)(ws + off); off += al((size_t)NN * 256 * 4);  // scatter accumulator
    float*  deg  = (float*)(ws + off); off += al((size_t)NN * 4);        // degrees

    hipMemsetAsync(deg, 0, (size_t)NN * 4, stream);
    hipMemsetAsync(agg, 0, (size_t)NN * 256 * 4, stream);

    // precision staging
    k_cvt_bf16<<<(NN * 256 / 4 + 255) / 256, 256, 0, stream>>>(x, xbf, NN * 256 / 4);
    k_pack_w<<<(512 * 64 + 255) / 256, 256, 0, stream>>>(Wl1, Wr1, 256, 512, wcb);
    k_degree<<<(E + 255) / 256, 256, 0, stream>>>(edst, deg, E);

    // layer 1: [Y|Z] = X @ [W_l1;W_r1]^T   (M=50000, N=512, K=256)
    dim3 g1((NN + 63) / 64, 512 / 64);
    k_gemm_bf16<<<g1, 128, 0, stream>>>(xbf, wcb, cbuf, NN, 512);

    // aggregate projected features (GEMM-first reassociation)
    k_scatter<256><<<(E * 32 + 255) / 256, 256, 0, stream>>>(cbuf, 512, esrc, edst, agg, E);
    k_finalize1<<<(NN * 64 + 255) / 256, 256, 0, stream>>>(agg, deg, b1, cbuf + 256, xbf, NN);

    // layer 2: [Y2|Z2] = H @ [W_l2;W_r2]^T  (M=50000, N=128, K=256)
    k_pack_w<<<(128 * 64 + 255) / 256, 256, 0, stream>>>(Wl2, Wr2, 64, 128, wcb);
    hipMemsetAsync(agg, 0, (size_t)NN * 64 * 4, stream);
    dim3 g2((NN + 63) / 64, 128 / 64);
    k_gemm_bf16<<<g2, 128, 0, stream>>>(xbf, wcb, cbuf, NN, 128);

    k_scatter<64><<<(E * 32 + 255) / 256, 256, 0, stream>>>(cbuf, 128, esrc, edst, agg, E);
    k_finalize2<<<(NN * 16 + 255) / 256, 256, 0, stream>>>(agg, deg, b2, cbuf + 64, out, NN);
}
